// RelKDAdapter_39633958208184
// MI455X (gfx1250) — compile-verified, matrix operand order checked by reading, not run on hardware
//
#include <hip/hip_runtime.h>

typedef float  v2f   __attribute__((ext_vector_type(2)));
typedef float  v8f   __attribute__((ext_vector_type(8)));
typedef __bf16 v16bf __attribute__((ext_vector_type(16)));

#define KDIM 256
#define NF   128
#define LDA  260   // LDS stride in floats: %4==0 (b128 align), %64==4 (bank spread)

// ---------------------------------------------------------------------------
// Projection GEMM: XP[16 rows x 128 cols per block] = X[16x256] @ W[128x256]^T
// Block = 256 threads = 8 waves; wave w computes the 16x16 tile at cols 16w.
// ---------------------------------------------------------------------------
__global__ __launch_bounds__(256) void proj_gemm(const float* __restrict__ X,
                                                 const float* __restrict__ W,
                                                 float* __restrict__ XP) {
    __shared__ float As[16 * LDA];
    const int tid      = threadIdx.x;
    const int row_base = blockIdx.x << 4;

    // Stage A tile (16 rows x 256 k) into LDS, fully coalesced float4 loads.
    {
        const int r  = tid >> 4;          // 0..15
        const int c0 = (tid & 15) << 4;   // 0,16,...,240
        const float4* g = (const float4*)(X + (size_t)(row_base + r) * KDIM + c0);
        float4* s = (float4*)(As + r * LDA + c0);
        s[0] = g[0]; s[1] = g[1]; s[2] = g[2]; s[3] = g[3];
    }
    __syncthreads();

    const int lane = tid & 31;
    const int colb = (tid >> 5) << 4;     // wave id * 16
    const int m    = lane & 15;           // A row / B col within tile
    v8f acc = {0.f, 0.f, 0.f, 0.f, 0.f, 0.f, 0.f, 0.f};

#if defined(__gfx1250__)
    const float* wrow = W + (size_t)(colb + m) * KDIM;   // B[k,n] = W[colb+n][k]
  #if __has_builtin(__builtin_amdgcn_wmma_f32_16x16x4_f32)
    const int kh = (lane >> 4) << 1;      // lanes 0-15: k,k+1 ; lanes 16-31: k+2,k+3
    #pragma unroll 8
    for (int k0 = 0; k0 < KDIM; k0 += 4) {
        const float* ap = As + m * LDA + k0 + kh;
        v2f a; a.x = ap[0]; a.y = ap[1];
        const float2 bw = *(const float2*)(wrow + k0 + kh);
        v2f b; b.x = bw.x; b.y = bw.y;
        acc = __builtin_amdgcn_wmma_f32_16x16x4_f32(false, a, false, b,
                                                    (short)0, acc, false, false);
    }
  #else
    // Fallback: bf16 WMMA (codegen-confirmed builtin); 16-bit A/B VGPR layout.
    const int half = lane >> 4;
    for (int k0 = 0; k0 < KDIM; k0 += 32) {
        v16bf a, b;
        #pragma unroll
        for (int e = 0; e < 16; ++e) {
            const int kk = k0 + ((e & 8) ? 16 : 0) + (half << 3) + (e & 7);
            a[e] = (__bf16)As[m * LDA + kk];
            b[e] = (__bf16)wrow[kk];
        }
        acc = __builtin_amdgcn_wmma_f32_16x16x32_bf16(false, a, false, b,
                                                      (short)0, acc, false, false);
    }
  #endif
#else
    (void)W; // host pass: never executed on device
#endif

    // C/D layout: lanes 0-15 -> N=lane, M=r ; lanes 16-31 -> N=lane-16, M=r+8
    const int row_off = (lane >> 4) << 3;
    float* outp = XP + (size_t)(row_base + row_off) * NF + colb + m;
    #pragma unroll
    for (int r = 0; r < 8; ++r) outp[(size_t)r * NF] = acc[r];
}

// ---------------------------------------------------------------------------
// Edge scatter: one wave per edge; lanes cover the 128 features in 4 coalesced
// 128-byte bursts of global_atomic_add_f32. Lane 0 bumps the degree counter.
// ---------------------------------------------------------------------------
__global__ __launch_bounds__(256) void edge_scatter(const int* __restrict__ src,
                                                    const int* __restrict__ dst,
                                                    const float* __restrict__ xp,
                                                    float* __restrict__ acc,
                                                    float* __restrict__ deg,
                                                    int E) {
    const int e = (blockIdx.x << 3) + (threadIdx.x >> 5);
    if (e >= E) return;
    const int lane = threadIdx.x & 31;
    const int s = src[e];
    const int d = dst[e];
    const float* xs = xp + (size_t)s * NF;
    float* ad       = acc + (size_t)d * NF;
    #pragma unroll
    for (int j = 0; j < 4; ++j) {
        const int f = lane + (j << 5);
        atomicAdd(ad + f, xs[f]);
    }
    if (lane == 0) atomicAdd(deg + d, 1.0f);
}

__global__ void finalize_mean(float* __restrict__ acc, const float* __restrict__ deg,
                              int n) {
    int i = blockIdx.x * blockDim.x + threadIdx.x;
    const int stride = gridDim.x * blockDim.x;
    for (; i < n; i += stride) {
        float dg = deg[i >> 7];           // row = i / 128
        dg = dg < 1.0f ? 1.0f : dg;
        acc[i] = acc[i] / dg;
    }
}

__global__ void zero_f32(float* __restrict__ p, long long n) {
    long long i = (long long)blockIdx.x * blockDim.x + threadIdx.x;
    const long long stride = (long long)gridDim.x * blockDim.x;
    for (; i < n; i += stride) p[i] = 0.0f;
}

// ---------------------------------------------------------------------------
extern "C" void kernel_launch(void* const* d_in, const int* in_sizes, int n_in,
                              void* d_out, int out_size, void* d_ws, size_t ws_size,
                              hipStream_t stream) {
    (void)n_in; (void)out_size; (void)ws_size;
    const float* x_a = (const float*)d_in[0];
    const float* x_b = (const float*)d_in[1];
    const float* W_a = (const float*)d_in[2];
    const float* W_b = (const float*)d_in[3];
    const int*   ei1 = (const int*)d_in[4];
    const int*   ei2 = (const int*)d_in[5];

    const int N = in_sizes[0] / KDIM;     // 100000 (divisible by 16)
    const int E = in_sizes[4] / 2;        // 1600000

    float* out = (float*)d_out;
    const size_t SZ = (size_t)N * NF;     // 12.8M floats per tensor
    float* dst1 = out;
    float* xp_a = out + SZ;
    float* dst2 = out + 2 * SZ;
    float* xp_b = out + 3 * SZ;
    float* deg1 = (float*)d_ws;
    float* deg2 = deg1 + N;

    // 1) zero accumulators (outputs poisoned once; ws not re-poisoned -> zero every call)
    zero_f32<<<2048, 256, 0, stream>>>(dst1, (long long)SZ);
    zero_f32<<<2048, 256, 0, stream>>>(dst2, (long long)SZ);
    zero_f32<<<128, 256, 0, stream>>>(deg1, N);
    zero_f32<<<128, 256, 0, stream>>>(deg2, N);

    // 2) per-type linear projections (WMMA GEMM)
    proj_gemm<<<N / 16, 256, 0, stream>>>(x_a, W_a, xp_a);
    proj_gemm<<<N / 16, 256, 0, stream>>>(x_b, W_b, xp_b);

    // 3) edge-type scatter-add + degree counting
    const int eb = (E + 7) / 8;
    edge_scatter<<<eb, 256, 0, stream>>>(ei1, ei1 + E, xp_a, dst1, deg1, E);
    edge_scatter<<<eb, 256, 0, stream>>>(ei2, ei2 + E, xp_b, dst2, deg2, E);

    // 4) divide by clamped degree (mean)
    finalize_mean<<<2048, 256, 0, stream>>>(dst1, deg1, (int)SZ);
    finalize_mean<<<2048, 256, 0, stream>>>(dst2, deg2, (int)SZ);
}
